// AttnProcessor_57939108823377
// MI455X (gfx1250) — matrix operand order repaired and checked by reading, not compiled, not run on hardware
//
#include <hip/hip_runtime.h>

// ---------------------------------------------------------------------------
// Types for CDNA5 WMMA (wave32, V_WMMA_F32_16X16X32_BF16)
// ---------------------------------------------------------------------------
typedef __bf16 v16bf __attribute__((ext_vector_type(16)));
typedef float  f32x8 __attribute__((ext_vector_type(8)));
typedef unsigned short ush;
typedef unsigned short ushx8 __attribute__((ext_vector_type(8)));
typedef unsigned short ushx4 __attribute__((ext_vector_type(4)));
typedef __attribute__((address_space(3))) ush lds_ush;

union Frag {
    v16bf v;
    ushx8 h[2];
};

__device__ __forceinline__ ush f2bf(float f) {
    unsigned int u = __float_as_uint(f);
    unsigned int r = u + 0x7FFFu + ((u >> 16) & 1u);   // round-to-nearest-even
    return (ush)(r >> 16);
}

// ---------------------------------------------------------------------------
// CDNA5 async global->LDS copy (GLOBAL_LOAD_ASYNC_TO_LDS_B128, ASYNCcnt).
// No VGPR data movement; completion via s_wait_asynccnt. GV addressing:
// vdst = per-lane LDS byte offset, vaddr = per-lane 64-bit global address.
// ---------------------------------------------------------------------------
__device__ __forceinline__ void async_copy16(ush* lptr, const ush* gptr) {
    unsigned loff = (unsigned)(unsigned long long)(lds_ush*)lptr;
    unsigned long long gaddr = (unsigned long long)gptr;
    asm volatile("global_load_async_to_lds_b128 %0, %1, off"
                 :: "v"(loff), "v"(gaddr) : "memory");
}
__device__ __forceinline__ void wait_async_le4() {
    asm volatile("s_wait_asynccnt 0x4" ::: "memory");
}
__device__ __forceinline__ void wait_async_0() {
    asm volatile("s_wait_asynccnt 0x0" ::: "memory");
}

// ---------------------------------------------------------------------------
// DPP16 butterfly reduction over 16-lane groups (pure VALU, co-executes with
// the XDL WMMA pipe).
// ---------------------------------------------------------------------------
template <int CTRL>
__device__ __forceinline__ float dpp_mov(float x) {
    return __int_as_float(__builtin_amdgcn_update_dpp(
        0, __float_as_int(x), CTRL, 0xF, 0xF, true));
}
__device__ __forceinline__ float row_max16(float x) {
    x = fmaxf(x, dpp_mov<0xB1>(x));    // quad_perm xor1
    x = fmaxf(x, dpp_mov<0x4E>(x));    // quad_perm xor2
    x = fmaxf(x, dpp_mov<0x141>(x));   // row_half_mirror
    x = fmaxf(x, dpp_mov<0x140>(x));   // row_mirror
    return x;
}
__device__ __forceinline__ float row_sum16(float x) {
    x += dpp_mov<0xB1>(x);
    x += dpp_mov<0x4E>(x);
    x += dpp_mov<0x141>(x);
    x += dpp_mov<0x140>(x);
    return x;
}

// ---------------------------------------------------------------------------
// f32 -> bf16 conversion (vectorized, 4 elems/thread)
// ---------------------------------------------------------------------------
__global__ __launch_bounds__(256) void cvt_f32_bf16(const float* __restrict__ in,
                                                    ush* __restrict__ out) {
    size_t i = ((size_t)blockIdx.x * 256 + threadIdx.x) * 4;
    float4 v = *(const float4*)(in + i);
    ushx4 r;
    r[0] = f2bf(v.x); r[1] = f2bf(v.y); r[2] = f2bf(v.z); r[3] = f2bf(v.w);
    *(ushx4*)(out + i) = r;
}

// ---------------------------------------------------------------------------
// f32 -> bf16 with transpose: out[n*dim + k] = bf16(in[k*dim + n]).
// ---------------------------------------------------------------------------
__global__ __launch_bounds__(256) void cvt_transpose_bf16(const float* __restrict__ in,
                                                          ush* __restrict__ out) {
    constexpr int DIM = 1024;
    __shared__ float tile[32][33];
    const int bx = blockIdx.x * 32, by = blockIdx.y * 32;
    const int tx = threadIdx.x & 31, ty4 = (threadIdx.x >> 5) * 4;
    #pragma unroll
    for (int i = 0; i < 4; ++i)
        tile[ty4 + i][tx] = in[(size_t)(by + ty4 + i) * DIM + bx + tx];
    __syncthreads();
    #pragma unroll
    for (int i = 0; i < 4; ++i)
        out[(size_t)(bx + ty4 + i) * DIM + by + tx] = f2bf(tile[tx][ty4 + i]);
}

// ---------------------------------------------------------------------------
// bf16 WMMA GEMM:  Y[M,1024] = A[M,1024] @ W[1024,1024] + bias, then *scale
//   Wt pre-transposed (Wt[n][k]). Double-buffered LDS fed by async copies.
// Block tile 128x128, BK=32, 256 threads = 8 waves, each wave 32x64 (2x4 WMMA).
// ---------------------------------------------------------------------------
template <bool FINAL, bool TRANSV>
__global__ __launch_bounds__(256) void gemm_bf16(const ush* __restrict__ A,
                                                 const ush* __restrict__ Wt,
                                                 const float* __restrict__ bias,
                                                 const float* __restrict__ resid,
                                                 ush* __restrict__ outBF,
                                                 float* __restrict__ outF,
                                                 float scale) {
    constexpr int Mtot = 4096, N = 1024, K = 1024;
    constexpr int BM = 128, BN = 128, BK = 32, LDT = BK + 8;
    __shared__ __align__(16) ush sA[2][BM * LDT];
    __shared__ __align__(16) ush sB[2][BN * LDT];

    const int tid  = threadIdx.x;
    const int lane = tid & 31;
    const int w    = tid >> 5;
    const int wr   = w >> 1;
    const int wc   = w & 1;
    const int l16  = lane & 15;
    const int lh   = lane >> 4;
    const int bm   = blockIdx.y * BM;
    const int bn   = blockIdx.x * BN;

    // Per-thread staging slots (4 async b128 per thread per tile)
    const int u0   = tid * 2;
    const int row0 = u0 >> 2,       d0 = (u0 & 3) * 8;
    const int row1 = (u0 + 1) >> 2, d1 = ((u0 + 1) & 3) * 8;

    auto issueTile = [&](int kc, int buf) {
        async_copy16(&sA[buf][row0 * LDT + d0], A  + (size_t)(bm + row0) * K + kc + d0);
        async_copy16(&sA[buf][row1 * LDT + d1], A  + (size_t)(bm + row1) * K + kc + d1);
        async_copy16(&sB[buf][row0 * LDT + d0], Wt + (size_t)(bn + row0) * K + kc + d0);
        async_copy16(&sB[buf][row1 * LDT + d1], Wt + (size_t)(bn + row1) * K + kc + d1);
    };

    f32x8 acc[2][4] = {};
    issueTile(0, 0);

    int ib = 0;
    for (int kc = 0; kc < K; kc += BK, ib ^= 1) {
        if (kc + BK < K) { issueTile(kc + BK, ib ^ 1); wait_async_le4(); }
        else             { wait_async_0(); }
        __syncthreads();                       // all waves' tile kc landed

        Frag aF[2];
        #pragma unroll
        for (int i = 0; i < 2; ++i) {
            const ush* p = sA[ib] + (wr * 32 + i * 16 + l16) * LDT + lh * 8;
            aF[i].h[0] = *(const ushx8*)p;
            aF[i].h[1] = *(const ushx8*)(p + 16);
        }
        #pragma unroll
        for (int j = 0; j < 4; ++j) {
            Frag bF;
            const ush* p = sB[ib] + (wc * 64 + j * 16 + l16) * LDT + lh * 8;
            bF.h[0] = *(const ushx8*)p;
            bF.h[1] = *(const ushx8*)(p + 16);
            #pragma unroll
            for (int i = 0; i < 2; ++i)
                acc[i][j] = __builtin_amdgcn_wmma_f32_16x16x32_bf16(
                    false, aF[i].v, false, bF.v, (short)0, acc[i][j], false, false);
        }
        __syncthreads();                       // safe to overwrite buf ib^1
    }

    #pragma unroll
    for (int i = 0; i < 2; ++i)
        #pragma unroll
        for (int j = 0; j < 4; ++j)
            #pragma unroll
            for (int r = 0; r < 8; ++r) {
                int row = bm + wr * 32 + i * 16 + r + lh * 8;
                int col = bn + wc * 64 + j * 16 + l16;
                float v = (acc[i][j][r] + bias[col]) * scale;
                if (FINAL)
                    outF[(size_t)row * N + col] = v + resid[(size_t)row * N + col];
                else if (TRANSV)
                    outBF[(size_t)col * Mtot + row] = f2bf(v);
                else
                    outBF[(size_t)row * N + col] = f2bf(v);
            }
}

// ---------------------------------------------------------------------------
// Flash attention: one block = 128 query rows of one (b,h); 8 waves, each wave
// owns 16 full rows (wave-local online softmax, DPP reductions, base-2 exp).
// Q pre-scaled by log2(e)/sqrt(D); V pre-transposed. Double-buffered K/V tiles
// fed by async global->LDS copies.
// ---------------------------------------------------------------------------
__global__ __launch_bounds__(256) void flash_attn(const ush* __restrict__ Q,
                                                  const ush* __restrict__ Kp,
                                                  const ush* __restrict__ Vt,
                                                  ush* __restrict__ O) {
    constexpr int S = 2048, C = 1024, Mtot = 4096, LDT = 72;
    __shared__ __align__(16) ush sK[2][64 * LDT];     // [key][d]
    __shared__ __align__(16) ush sV[2][64 * LDT];     // [d][key]
    __shared__ __align__(16) ush sP[8 * 16 * LDT];    // per-wave P scratch

    const int tid  = threadIdx.x;
    const int lane = tid & 31;
    const int w    = tid >> 5;
    const int l16  = lane & 15;
    const int lh   = lane >> 4;
    const int q0   = blockIdx.x * 128;
    const int h    = blockIdx.y;
    const int b    = blockIdx.z;
    const size_t rowBase = (size_t)b * S;
    const int colBase = h * 64;

    // Per-thread staging slots (2 K + 2 V async b128 per thread per tile)
    const int u0 = tid * 2;
    const int r80 = u0 >> 3,       c80 = (u0 & 7) * 8;
    const int r81 = (u0 + 1) >> 3, c81 = ((u0 + 1) & 7) * 8;

    auto issueKV = [&](int kv, int buf) {
        async_copy16(&sK[buf][r80 * LDT + c80],
                     Kp + (rowBase + kv + r80) * C + colBase + c80);
        async_copy16(&sK[buf][r81 * LDT + c81],
                     Kp + (rowBase + kv + r81) * C + colBase + c81);
        async_copy16(&sV[buf][r80 * LDT + c80],
                     Vt + (size_t)(colBase + r80) * Mtot + rowBase + kv + c80);
        async_copy16(&sV[buf][r81 * LDT + c81],
                     Vt + (size_t)(colBase + r81) * Mtot + rowBase + kv + c81);
    };

    Frag qF[2];
    {
        const ush* qp = Q + (rowBase + q0 + w * 16 + l16) * C + colBase;
        #pragma unroll
        for (int f = 0; f < 2; ++f) {
            int d0 = f * 32 + lh * 8;
            qF[f].h[0] = *(const ushx8*)(qp + d0);
            qF[f].h[1] = *(const ushx8*)(qp + d0 + 16);
        }
    }

    float m[8], lsum[8];
    f32x8 o[4] = {};
    #pragma unroll
    for (int r = 0; r < 8; ++r) { m[r] = -1e30f; lsum[r] = 0.f; }

    ush* myP = sP + w * (16 * LDT);
    issueKV(0, 0);

    int ib = 0;
    for (int kv = 0; kv < S; kv += 64, ib ^= 1) {
        if (kv + 64 < S) { issueKV(kv + 64, ib ^ 1); wait_async_le4(); }
        else             { wait_async_0(); }
        __syncthreads();                       // all waves' K/V tile landed

        // Scores: 16x64 = Q(16x64) @ K^T, 8 WMMAs (logits already in log2 dom.)
        f32x8 s[4] = {};
        #pragma unroll
        for (int kk = 0; kk < 2; ++kk)
            #pragma unroll
            for (int t = 0; t < 4; ++t) {
                Frag bK;
                const ush* p = sK[ib] + (t * 16 + l16) * LDT + kk * 32 + lh * 8;
                bK.h[0] = *(const ushx8*)p;
                bK.h[1] = *(const ushx8*)(p + 16);
                s[t] = __builtin_amdgcn_wmma_f32_16x16x32_bf16(
                    false, qF[kk].v, false, bK.v, (short)0, s[t], false, false);
            }

        // Online softmax in base 2 (v_exp_f32 direct), DPP row reductions
        #pragma unroll
        for (int r = 0; r < 8; ++r) {
            float tmax = fmaxf(fmaxf(s[0][r], s[1][r]), fmaxf(s[2][r], s[3][r]));
            tmax = row_max16(tmax);
            float mnew = fmaxf(m[r], tmax);
            float corr = __builtin_amdgcn_exp2f(m[r] - mnew);
            m[r] = mnew;
            lsum[r] *= corr;
            #pragma unroll
            for (int c = 0; c < 4; ++c) o[c][r] *= corr;
            float ps = 0.f;
            #pragma unroll
            for (int t = 0; t < 4; ++t) {
                float p = __builtin_amdgcn_exp2f(s[t][r] - mnew);
                s[t][r] = p;
                ps += p;
            }
            lsum[r] += row_sum16(ps);
        }

        // Re-layout P (C-format -> A-format) through wave-private LDS
        #pragma unroll
        for (int t = 0; t < 4; ++t)
            #pragma unroll
            for (int r = 0; r < 8; ++r)
                myP[(r + lh * 8) * LDT + t * 16 + l16] = f2bf(s[t][r]);

        // O += P(16x64) @ V(64x64), 8 WMMAs
        #pragma unroll
        for (int kk = 0; kk < 2; ++kk) {
            Frag pA;
            const ush* pp = myP + l16 * LDT + kk * 32 + lh * 8;
            pA.h[0] = *(const ushx8*)pp;
            pA.h[1] = *(const ushx8*)(pp + 16);
            #pragma unroll
            for (int c = 0; c < 4; ++c) {
                Frag bV;
                const ush* vp = sV[ib] + (c * 16 + l16) * LDT + kk * 32 + lh * 8;
                bV.h[0] = *(const ushx8*)vp;
                bV.h[1] = *(const ushx8*)(vp + 16);
                o[c] = __builtin_amdgcn_wmma_f32_16x16x32_bf16(
                    false, pA.v, false, bV.v, (short)0, o[c], false, false);
            }
        }
        __syncthreads();                       // safe to overwrite buf ib^1
    }

    #pragma unroll
    for (int r = 0; r < 8; ++r) {
        float inv = 1.f / lsum[r];
        size_t row = rowBase + q0 + w * 16 + r + lh * 8;
        #pragma unroll
        for (int c = 0; c < 4; ++c)
            O[row * C + colBase + c * 16 + l16] = f2bf(o[c][r] * inv);
    }
}

// ---------------------------------------------------------------------------
// Launch: convert/transpose -> QKV GEMMs -> flash attention -> out-proj
// ---------------------------------------------------------------------------
extern "C" void kernel_launch(void* const* d_in, const int* in_sizes, int n_in,
                              void* d_out, int out_size, void* d_ws, size_t ws_size,
                              hipStream_t stream) {
    const float* X  = (const float*)d_in[0];
    const float* Wq = (const float*)d_in[1];
    const float* bq = (const float*)d_in[2];
    const float* Wk = (const float*)d_in[3];
    const float* bk = (const float*)d_in[4];
    const float* Wv = (const float*)d_in[5];
    const float* bv = (const float*)d_in[6];
    const float* Wo = (const float*)d_in[7];
    const float* bo = (const float*)d_in[8];
    float* out = (float*)d_out;

    const size_t M = 2 * 2048;   // B*S
    const size_t C = 1024;

    char* p = (char*)d_ws;
    ush* Xh  = (ush*)p; p += M * C * 2;
    ush* Wqt = (ush*)p; p += C * C * 2;
    ush* Wkt = (ush*)p; p += C * C * 2;
    ush* Wvt = (ush*)p; p += C * C * 2;
    ush* Wot = (ush*)p; p += C * C * 2;
    ush* Qh  = (ush*)p; p += M * C * 2;   // pre-scaled by log2(e)/8
    ush* Kh  = (ush*)p; p += M * C * 2;
    ush* Vtt = (ush*)p; p += M * C * 2;   // transposed Vt[C][M]
    ush* AOh = (ush*)p; p += M * C * 2;

    cvt_f32_bf16<<<(int)(M * C / 1024), 256, 0, stream>>>(X, Xh);
    dim3 gT(32, 32);
    cvt_transpose_bf16<<<gT, 256, 0, stream>>>(Wq, Wqt);
    cvt_transpose_bf16<<<gT, 256, 0, stream>>>(Wk, Wkt);
    cvt_transpose_bf16<<<gT, 256, 0, stream>>>(Wv, Wvt);
    cvt_transpose_bf16<<<gT, 256, 0, stream>>>(Wo, Wot);

    const float kQScale = 0.125f * 1.44269504088896340736f;  // log2(e)/sqrt(D)
    dim3 gGemm(C / 128, M / 128);            // (8, 32)
    gemm_bf16<false, false><<<gGemm, 256, 0, stream>>>(Xh, Wqt, bq, nullptr, Qh,  nullptr, kQScale);
    gemm_bf16<false, false><<<gGemm, 256, 0, stream>>>(Xh, Wkt, bk, nullptr, Kh,  nullptr, 1.0f);
    gemm_bf16<false, true ><<<gGemm, 256, 0, stream>>>(Xh, Wvt, bv, nullptr, Vtt, nullptr, 1.0f);

    flash_attn<<<dim3(16, 16, 2), 256, 0, stream>>>(Qh, Kh, Vtt, AOh);

    gemm_bf16<true, false><<<gGemm, 256, 0, stream>>>(AOh, Wot, bo, X, nullptr, out, 1.0f);
}
